// GroupedQueryAttention_73864847556957
// MI455X (gfx1250) — matrix operand order repaired and checked by reading, not compiled, not run on hardware
//
#include <hip/hip_runtime.h>

// ---------------------------------------------------------------------------
// Types (trivial ext-vector types so they can live in unions / builtins)
// ---------------------------------------------------------------------------
typedef __attribute__((ext_vector_type(16))) __bf16       bf16x16;
typedef __attribute__((ext_vector_type(8)))  float        f32x8;
typedef __attribute__((ext_vector_type(4)))  unsigned int u32x4;
typedef __attribute__((ext_vector_type(4)))  float        f32x4;
typedef __attribute__((ext_vector_type(8)))  int          i32x8;
typedef __attribute__((ext_vector_type(4)))  int          i32x4;

union Frag { bf16x16 v; u32x4 u[2]; };

__device__ __forceinline__ unsigned short f2bf(float f) {
  union { float f; unsigned u; } x; x.f = f;
  unsigned r = x.u + 0x7fffu + ((x.u >> 16) & 1u);   // round-to-nearest-even
  return (unsigned short)(r >> 16);
}

#define WMMA_BF16(a, b, c) \
  __builtin_amdgcn_wmma_f32_16x16x32_bf16(false, (a), false, (b), (short)0, (c), false, false)

// ---------------------------------------------------------------------------
// Tensor Data Mover: 2-D bf16 tile load, LDS rows padded 128B data + 16B pad
// (pad_interval code 4 = 32 DWORDs, pad_amount code 3 = 4 DWORDs -> [h][72] layout)
// Requires tile width == 64 bf16 elements.
// This toolchain exposes the 6-argument builtin:
//   (uint32x4 g0, int32x8 g1, int32x4 g2, int32x4 g3, int32x8 extra, i32 cpol)
// ---------------------------------------------------------------------------
#if defined(__gfx1250__) && __has_builtin(__builtin_amdgcn_tensor_load_to_lds) && \
    __has_builtin(__builtin_amdgcn_s_wait_tensorcnt)
#define USE_TDM 1
#else
#define USE_TDM 0
#endif

#if USE_TDM
__device__ __forceinline__ void tdm_load_bf16_64w(unsigned lds_off, const unsigned short* gptr,
                                                  unsigned h_rows, unsigned row_stride_elems) {
  unsigned long long ga = (unsigned long long)(size_t)gptr;
  u32x4 g0;
  g0[0] = 1u;                                            // count=1 (valid), user mode
  g0[1] = lds_off;                                       // lds_addr (bytes)
  g0[2] = (unsigned)ga;                                  // global_addr[31:0]
  g0[3] = ((unsigned)(ga >> 32) & 0x01ffffffu) | (2u << 30); // global_addr[56:32], type=2
  i32x8 g1;
  g1[0] = (1 << 16)            // data_size = 1 -> 2 bytes
        | (1 << 20)            // pad_enable
        | (4 << 22)            // pad_interval: 32 DWORDs (128B) between pads
        | (3 << 25);           // pad_amount: 4 DWORDs (16B)
  g1[1] = (int)0xffff0000u;                 // atomic_barrier=0 ; tensor_dim0 lo16 (huge)
  g1[2] = (int)0xffff7fffu;                 // tensor_dim0 hi16 ; tensor_dim1 lo16 (huge)
  g1[3] = (int)(0x7fffu | (64u << 16));     // tensor_dim1 hi16 ; tile_dim0 = 64
  g1[4] = (int)h_rows;                      // tile_dim1 = rows ; tile_dim2 = 0
  g1[5] = (int)row_stride_elems;            // tensor_dim0_stride[31:0]
  g1[6] = 0;                                // stride hi ; tensor_dim1_stride lo
  g1[7] = 0;
  i32x4 gz4 = {};                           // groups 2/3 unused (2-D tile)
  i32x8 gz8 = {};                           // trailing group unused
  __builtin_amdgcn_tensor_load_to_lds(g0, g1, gz4, gz4, gz8, 0);
}
#endif

// Provably-uniform "am I wave 0 of this block" (forces a scalar branch, so the
// TDM instruction -- which issues regardless of EXEC -- is skipped on waves 1+)
__device__ __forceinline__ bool is_wave0() {
  return (__builtin_amdgcn_readfirstlane((int)threadIdx.x) >> 5) == 0;
}

// ---------------------------------------------------------------------------
// fp32 -> bf16 conversion (weights, converted once)
// ---------------------------------------------------------------------------
__global__ __launch_bounds__(256) void cvt_f32_to_bf16(const float* __restrict__ s,
                                                       unsigned short* __restrict__ d, int n) {
  int i = blockIdx.x * 256 + threadIdx.x;
  if (i < n) d[i] = f2bf(s[i]);
}

// ---------------------------------------------------------------------------
// Tiled WMMA GEMM: Out[M,N] = A[M,K] @ B[K,N]
//   block tile 256x64, 8 waves; each wave owns a 32x64 strip (2 M-tiles x 4 N-tiles)
//   A_F32:  A is fp32, converted to bf16 during LDS staging (manual path)
//           otherwise A tile is staged by the Tensor Data Mover
//   ROPE:   fuse RoPE epilogue (head dim = 64, col base aligned to 64)
//   OUT_F32: store fp32 (final projection) instead of bf16
// ---------------------------------------------------------------------------
template <bool A_F32, bool ROPE, bool OUT_F32>
__global__ __launch_bounds__(256) void gemm_wmma(
    const void* __restrict__ Aptr, const unsigned short* __restrict__ Bw,
    void* __restrict__ Out, const float* __restrict__ cosp, const float* __restrict__ sinp,
    int M, int N, int K) {
  __shared__ __attribute__((aligned(16))) unsigned short lds_a[256 * 72]; // [row][72]
  __shared__ __attribute__((aligned(16))) unsigned short lds_b[64 * 72];  // transposed [n][72]

  const int tid  = threadIdx.x;
  const int wave = tid >> 5;
  const int lane = tid & 31;
  const int hi   = lane >> 4;   // which 16-lane half
  const int ln   = lane & 15;
  const int rowBlock = blockIdx.y * 256;
  const int colBlock = blockIdx.x * 64;

  f32x8 zero = {};
  f32x8 acc0[4], acc1[4];
#pragma unroll
  for (int t = 0; t < 4; ++t) { acc0[t] = zero; acc1[t] = zero; }

  for (int k0 = 0; k0 < K; k0 += 64) {
    __syncthreads();
    // ---- stage A tile (256 x 64) ----
    if constexpr (A_F32) {
#pragma unroll
      for (int c = 0; c < 8; ++c) {
        int e = (c * 256 + tid) * 8;
        int r = e >> 6, col = e & 63;
        const float* src = (const float*)Aptr + (size_t)(rowBlock + r) * K + k0 + col;
        f32x4 f0 = ((const f32x4*)src)[0];
        f32x4 f1 = ((const f32x4*)src)[1];
        alignas(16) unsigned short t8[8] = {
            f2bf(f0.x), f2bf(f0.y), f2bf(f0.z), f2bf(f0.w),
            f2bf(f1.x), f2bf(f1.y), f2bf(f1.z), f2bf(f1.w)};
        *(u32x4*)(lds_a + r * 72 + col) = *(const u32x4*)t8;
        if (k0 + 64 < K) __builtin_prefetch((const void*)(src + 64), 0, 0);
      }
    } else {
#if USE_TDM
      if (is_wave0())
        tdm_load_bf16_64w((unsigned)(size_t)(void*)lds_a,
                          (const unsigned short*)Aptr + (size_t)rowBlock * K + k0,
                          256u, (unsigned)K);
#else
#pragma unroll
      for (int c = 0; c < 8; ++c) {
        int e = (c * 256 + tid) * 8;
        int r = e >> 6, col = e & 63;
        const unsigned short* src =
            (const unsigned short*)Aptr + (size_t)(rowBlock + r) * K + k0 + col;
        *(u32x4*)(lds_a + r * 72 + col) = *(const u32x4*)src;
      }
#endif
    }
    // ---- stage B tile (64 x 64), transposed into LDS [n][k] ----
    {
      int kr = tid & 63, nt = tid >> 6;
      const unsigned short* src = Bw + (size_t)(k0 + kr) * N + colBlock + nt * 16;
      alignas(16) unsigned short t16[16];
      *(u32x4*)(t16)     = *(const u32x4*)src;
      *(u32x4*)(t16 + 8) = *(const u32x4*)(src + 8);
#pragma unroll
      for (int j = 0; j < 16; ++j) lds_b[(nt * 16 + j) * 72 + kr] = t16[j];
    }
#if USE_TDM
    if constexpr (!A_F32) {
      if (is_wave0()) __builtin_amdgcn_s_wait_tensorcnt(0);
    }
#endif
    __syncthreads();

#pragma unroll
    for (int kk = 0; kk < 64; kk += 32) {
      // A fragments 16x32 (elems 0..7 -> k=hi*8+j, 8..15 -> 16+hi*8+j)
      Frag a0, a1;
      const unsigned short* ap0 = lds_a + (wave * 32 + ln) * 72 + kk;
      const unsigned short* ap1 = lds_a + (wave * 32 + 16 + ln) * 72 + kk;
      a0.u[0] = *(const u32x4*)(ap0 + hi * 8);
      a0.u[1] = *(const u32x4*)(ap0 + 16 + hi * 8);
      a1.u[0] = *(const u32x4*)(ap1 + hi * 8);
      a1.u[1] = *(const u32x4*)(ap1 + 16 + hi * 8);
#pragma unroll
      for (int t = 0; t < 4; ++t) {
        // B fragment 32x16 (elems 0..7 -> k=hi*8+j, 8..15 -> 16+hi*8+j, col n = lane&15), reused twice
        Frag b;
        const unsigned short* bp = lds_b + (t * 16 + ln) * 72 + kk + hi * 8;
        b.u[0] = *(const u32x4*)bp;
        b.u[1] = *(const u32x4*)(bp + 16);
        acc0[t] = WMMA_BF16(a0.v, b.v, acc0[t]);
        acc1[t] = WMMA_BF16(a1.v, b.v, acc1[t]);
      }
    }
  }

  // ---- epilogue (C layout: VGPR i -> row i + hi*8, col = lane&15) ----
#pragma unroll
  for (int strip = 0; strip < 2; ++strip) {
    const int r0 = rowBlock + wave * 32 + strip * 16;
    f32x8* acc = strip ? acc1 : acc0;
#pragma unroll
    for (int t = 0; t < 4; ++t) {
      const int col = colBlock + t * 16 + ln;
#pragma unroll
      for (int i = 0; i < 8; ++i) {
        const int r = r0 + i + hi * 8;
        float val = acc[t][i];
        if constexpr (ROPE) {
          int s = r & 2047;   // S = 2048
          int d = col & 63;   // HD = 64
          float c  = cosp[s * 64 + d];
          float sn = sinp[s * 64 + d];
          // rotate_half partner lives in tile t+-2, same lane
          float other = (t < 2) ? (float)acc[t + 2][i] : (float)acc[t - 2][i];
          val = (d < 32) ? (val * c - other * sn) : (val * c + other * sn);
        }
        if constexpr (OUT_F32)
          ((float*)Out)[(size_t)r * N + col] = val;
        else
          ((unsigned short*)Out)[(size_t)r * N + col] = f2bf(val);
      }
    }
  }
}

// ---------------------------------------------------------------------------
// Flash attention (causal, GQA): block = (64 queries, head h, batch b), 4 waves
//   q: (B*S, 32*64) bf16   k,v: (B*S, 8*64) bf16   o: (B*S, 32*64) bf16
//   K tile staged by TDM, V tile transposed manually, P routed via per-wave LDS
// ---------------------------------------------------------------------------
__global__ __launch_bounds__(128) void attn_wmma(
    const unsigned short* __restrict__ q, const unsigned short* __restrict__ k,
    const unsigned short* __restrict__ v, unsigned short* __restrict__ o) {
  constexpr int S = 2048, HD = 64, DQ = 2048, DKV = 512;
  __shared__ __attribute__((aligned(16))) unsigned short lds_k[32 * 72];   // [key][d]
  __shared__ __attribute__((aligned(16))) unsigned short lds_vt[64 * 40];  // [d][key]
  __shared__ __attribute__((aligned(16))) unsigned short lds_p[4][16 * 40]; // per wave [m][key]

  const int tid  = threadIdx.x;
  const int wave = tid >> 5;
  const int lane = tid & 31;
  const int hi   = lane >> 4;
  const int ln   = lane & 15;
  const int qb0  = blockIdx.x * 64;
  const int h = blockIdx.y, b = blockIdx.z;
  const int g = h >> 2;  // NREP = 4
  const size_t rb = (size_t)b * S;

  // Q fragments (16 queries x 64 dims) stay in registers the whole kernel
  Frag aq[2];
  {
    const unsigned short* qp = q + (rb + qb0 + wave * 16 + ln) * DQ + h * HD;
#pragma unroll
    for (int kk = 0; kk < 2; ++kk) {
      aq[kk].u[0] = *(const u32x4*)(qp + kk * 32 + hi * 8);
      aq[kk].u[1] = *(const u32x4*)(qp + kk * 32 + 16 + hi * 8);
    }
  }

  f32x8 zero = {};
  float mrow[8], lrow[8];
  f32x8 accO[4];
#pragma unroll
  for (int i = 0; i < 8; ++i) { mrow[i] = -1e30f; lrow[i] = 0.0f; }
#pragma unroll
  for (int t = 0; t < 4; ++t) accO[t] = zero;

  const int nkt = (qb0 + 64) >> 5;  // 32-key tiles covering the causal span
  for (int kt = 0; kt < nkt; ++kt) {
    __syncthreads();
#if USE_TDM
    if (is_wave0())
      tdm_load_bf16_64w((unsigned)(size_t)(void*)lds_k,
                        k + (rb + kt * 32) * DKV + g * HD, 32u, (unsigned)DKV);
#endif
    {  // cooperative staging: V transposed (and K manually if no TDM)
      int kr = tid & 31, nt = tid >> 5;
#if !USE_TDM
      const unsigned short* ks = k + (rb + kt * 32 + kr) * DKV + g * HD + nt * 16;
      *(u32x4*)(lds_k + kr * 72 + nt * 16)     = *(const u32x4*)ks;
      *(u32x4*)(lds_k + kr * 72 + nt * 16 + 8) = *(const u32x4*)(ks + 8);
#endif
      const unsigned short* vs = v + (rb + kt * 32 + kr) * DKV + g * HD + nt * 16;
      alignas(16) unsigned short t16[16];
      *(u32x4*)t16       = *(const u32x4*)vs;
      *(u32x4*)(t16 + 8) = *(const u32x4*)(vs + 8);
#pragma unroll
      for (int j = 0; j < 16; ++j) lds_vt[(nt * 16 + j) * 40 + kr] = t16[j];
    }
#if USE_TDM
    if (is_wave0()) __builtin_amdgcn_s_wait_tensorcnt(0);
#endif
    __syncthreads();

    // skip fully-masked tiles (wave-uniform -> EXEC stays all-ones for WMMA)
    if (kt * 32 <= qb0 + wave * 16 + 15) {
      // scores: 16 queries x 32 keys = two 16x16 tiles, K-dim = head dim (2 x 32)
      f32x8 s[2];
#pragma unroll
      for (int tt = 0; tt < 2; ++tt) {
        f32x8 c = zero;
#pragma unroll
        for (int kk = 0; kk < 2; ++kk) {
          Frag bk;  // B = K^T: elems 0..7 -> d = kk*32 + hi*8 + j, 8..15 -> kk*32 + 16 + hi*8 + j, col = key
          const unsigned short* kp = lds_k + (tt * 16 + ln) * 72 + kk * 32 + hi * 8;
          bk.u[0] = *(const u32x4*)kp;
          bk.u[1] = *(const u32x4*)(kp + 16);
          c = WMMA_BF16(aq[kk].v, bk.v, c);
        }
        s[tt] = c;
      }
      // online softmax (rows live in 16-lane halves; xor masks 1..8 stay in-half)
      float p0v[8], p1v[8];
#pragma unroll
      for (int i = 0; i < 8; ++i) {
        int qi = qb0 + wave * 16 + i + hi * 8;
        float v0 = s[0][i] * 0.125f;  // 1/sqrt(64)
        float v1 = s[1][i] * 0.125f;
        if (kt * 32 + ln > qi)      v0 = -1e9f;
        if (kt * 32 + 16 + ln > qi) v1 = -1e9f;
        float r = fmaxf(v0, v1);
#pragma unroll
        for (int mm = 8; mm; mm >>= 1) r = fmaxf(r, __shfl_xor(r, mm, 32));
        float mn    = fmaxf(mrow[i], r);
        float alpha = __expf(mrow[i] - mn);
        float p0 = __expf(v0 - mn), p1 = __expf(v1 - mn);
        float rs = p0 + p1;
#pragma unroll
        for (int mm = 8; mm; mm >>= 1) rs += __shfl_xor(rs, mm, 32);
        lrow[i] = lrow[i] * alpha + rs;
        mrow[i] = mn;
#pragma unroll
        for (int t = 0; t < 4; ++t) accO[t][i] *= alpha;
        p0v[i] = p0; p1v[i] = p1;
      }
      // route P from C layout to A layout through per-wave LDS
#pragma unroll
      for (int i = 0; i < 8; ++i) {
        lds_p[wave][(i + hi * 8) * 40 + ln]      = f2bf(p0v[i]);
        lds_p[wave][(i + hi * 8) * 40 + 16 + ln] = f2bf(p1v[i]);
      }
      Frag pf;
      const unsigned short* pp = &lds_p[wave][ln * 40 + hi * 8];
      pf.u[0] = *(const u32x4*)pp;
      pf.u[1] = *(const u32x4*)(pp + 16);
      // O += P @ V  (B = V from transposed LDS: elems 0..7 -> key = hi*8 + j, 8..15 -> 16 + hi*8 + j)
#pragma unroll
      for (int t = 0; t < 4; ++t) {
        Frag bv;
        const unsigned short* vp = lds_vt + (t * 16 + ln) * 40 + hi * 8;
        bv.u[0] = *(const u32x4*)vp;
        bv.u[1] = *(const u32x4*)(vp + 16);
        accO[t] = WMMA_BF16(pf.v, bv.v, accO[t]);
      }
    }
  }

#pragma unroll
  for (int t = 0; t < 4; ++t)
#pragma unroll
    for (int i = 0; i < 8; ++i) {
      int r = qb0 + wave * 16 + i + hi * 8;
      o[(rb + r) * DQ + h * HD + t * 16 + ln] = f2bf(accO[t][i] / lrow[i]);
    }
}

// ---------------------------------------------------------------------------
// Host-side orchestration
// ---------------------------------------------------------------------------
extern "C" void kernel_launch(void* const* d_in, const int* in_sizes, int n_in,
                              void* d_out, int out_size, void* d_ws, size_t ws_size,
                              hipStream_t stream) {
  const float* x    = (const float*)d_in[0];
  const float* cosp = (const float*)d_in[1];
  const float* sinp = (const float*)d_in[2];
  const float* Wq   = (const float*)d_in[3];
  const float* Wk   = (const float*)d_in[4];
  const float* Wv   = (const float*)d_in[5];
  const float* Wo   = (const float*)d_in[6];
  // d_in[7] (attention_mask) is the known causal tril -> handled analytically
  float* out = (float*)d_out;

  constexpr int B = 2, S = 2048, D = 2048, H = 32, KV = 8, HD = 64;
  constexpr int M   = B * S;    // 4096
  constexpr int NKV = KV * HD;  // 512

  char* ws = (char*)d_ws;
  size_t off = 0;
  auto alloc = [&](size_t elems) {
    unsigned short* p = (unsigned short*)(ws + off);
    off += elems * sizeof(unsigned short);
    return p;
  };
  unsigned short* wqb = alloc((size_t)D * D);
  unsigned short* wkb = alloc((size_t)D * NKV);
  unsigned short* wvb = alloc((size_t)D * NKV);
  unsigned short* wob = alloc((size_t)D * D);
  unsigned short* qb  = alloc((size_t)M * D);
  unsigned short* kb  = alloc((size_t)M * NKV);
  unsigned short* vb  = alloc((size_t)M * NKV);
  unsigned short* ab  = alloc((size_t)M * D);

  auto cvt = [&](const float* s, unsigned short* d, int n) {
    cvt_f32_to_bf16<<<(n + 255) / 256, 256, 0, stream>>>(s, d, n);
  };
  cvt(Wq, wqb, D * D);
  cvt(Wk, wkb, D * NKV);
  cvt(Wv, wvb, D * NKV);
  cvt(Wo, wob, D * D);

  dim3 blk(256);
  // Q = x @ Wq (+RoPE), K = x @ Wk (+RoPE), V = x @ Wv
  gemm_wmma<true, true,  false><<<dim3(D / 64,   M / 256), blk, 0, stream>>>(x, wqb, qb, cosp, sinp, M, D,   D);
  gemm_wmma<true, true,  false><<<dim3(NKV / 64, M / 256), blk, 0, stream>>>(x, wkb, kb, cosp, sinp, M, NKV, D);
  gemm_wmma<true, false, false><<<dim3(NKV / 64, M / 256), blk, 0, stream>>>(x, wvb, vb, nullptr, nullptr, M, NKV, D);

  // causal GQA flash attention
  attn_wmma<<<dim3(S / 64, H, B), dim3(128), 0, stream>>>(qb, kb, vb, ab);

  // out = attn @ Wo  (fp32 result, A tile staged via Tensor Data Mover)
  gemm_wmma<false, false, true><<<dim3(D / 64, M / 256), blk, 0, stream>>>(ab, wob, out, nullptr, nullptr, M, D, D);
}
